// PoolHiddenNet_82566451298929
// MI455X (gfx1250) — matrix operand order repaired and checked by reading, not compiled, not run on hardware
//
#include <hip/hip_runtime.h>

typedef __attribute__((ext_vector_type(16))) _Float16 v16h;
typedef __attribute__((ext_vector_type(8)))  _Float16 v8h;
typedef __attribute__((ext_vector_type(4)))  _Float16 v4h;
typedef __attribute__((ext_vector_type(8)))  float    v8f;

#define N_TOT  102400
#define GROUPS 50
#define NGRP   2048
#define E_DIM  128
#define H_DIM  512
#define K_TOT  640
#define NSTEP  (K_TOT / TILE_K)

#define TILE_M 128
#define TILE_N 128
#define TILE_K 32
#define LDA    40     // f16 k-stride for ldsA rows / ldsB cols (bank-conflict-free)
#define LDP    129    // f32 row stride for pooled staging

__global__ void init_out_kernel(float* out) {
    int i = blockIdx.x * blockDim.x + threadIdx.x;
    if (i < NGRP * H_DIM) out[i] = 0.0f;
}

__global__ __launch_bounds__(256)
void pool_gemm_kernel(const float* __restrict__ hist_enc,
                      const float* __restrict__ pos,
                      const int*   __restrict__ seq_start_end,
                      const float* __restrict__ Ws,
                      const float* __restrict__ bs,
                      const float* __restrict__ Wm,
                      const float* __restrict__ bm,
                      float* __restrict__ out) {
    __shared__ _Float16 ldsA[2][TILE_M * LDA];   // A tile: [row][k]  (f16), double buffered
    __shared__ _Float16 ldsB[2][TILE_N * LDA];   // B tile: [col][k]  (f16, transposed), double buffered
    __shared__ float    relS[TILE_M * 2];        // per-row relative position
    __shared__ float    pooled[TILE_M * LDP];    // post-ReLU tile for segment-max

    const int tid  = threadIdx.x;
    const int lane = tid & 31;
    const int l15  = lane & 15;
    const int hi   = lane >> 4;
    const int wave = tid >> 5;
    const int mGrp = wave & 3;    // 4 row-groups of 32 rows
    const int nGrp = wave >> 2;   // 2 col-groups of 64 cols

    const int c0 = blockIdx.x * TILE_N;
    const int r0 = blockIdx.y * TILE_M;

    // ---- per-thread fill coordinates ----
    const int aRow = tid >> 3;          // + 32*q  -> row within tile
    const int aK   = (tid & 7) << 2;    // k quad within tile
    const int bCol = tid & 127;         // column within tile (K-gather fill)
    const int bKh  = (tid >> 7) << 4;   // k half: 0 or 16

    // relative positions for this 128-row block
    if (tid < TILE_M) {
        int gr  = r0 + tid;
        int seg = gr / GROUPS;
        int st  = seq_start_end[seg];
        relS[2*tid+0] = pos[2*gr+0] - pos[2*st+0];
        relS[2*tid+1] = pos[2*gr+1] - pos[2*st+1];
    }

    v8f acc[2][4];
    for (int mt = 0; mt < 2; ++mt)
        for (int nt = 0; nt < 4; ++nt)
            for (int i = 0; i < 8; ++i) acc[mt][nt][i] = 0.0f;

    float4 aReg[4];       // prefetched A quads (hist_enc part)
    float  bRegF[16];     // prefetched B column segment (16 k-values, one column)

    // ---------------- prologue: load + store tile for k0 = 0 ----------------
    for (int q = 0; q < 4; ++q)
        aReg[q] = *(const float4*)(hist_enc + (size_t)(r0 + aRow + 32*q) * H_DIM + aK);
    {
        const float* bp = Wm + (size_t)bKh * H_DIM + c0 + bCol;
        for (int i = 0; i < 16; ++i) bRegF[i] = bp[(size_t)i * H_DIM];
    }
    for (int q = 0; q < 4; ++q) {
        v4h pk; pk[0] = (_Float16)aReg[q].x; pk[1] = (_Float16)aReg[q].y;
                pk[2] = (_Float16)aReg[q].z; pk[3] = (_Float16)aReg[q].w;
        *(v4h*)&ldsA[0][(aRow + 32*q) * LDA + aK] = pk;
    }
    for (int i2 = 0; i2 < 4; ++i2) {
        v4h pk;
        for (int j = 0; j < 4; ++j) pk[j] = (_Float16)bRegF[i2*4 + j];
        *(v4h*)&ldsB[0][bCol * LDA + bKh + i2*4] = pk;
    }
    __syncthreads();

    // ---------------- main pipelined loop ----------------
    for (int step = 0; step < NSTEP; ++step) {
        const int cur = step & 1;
        const int nxt = cur ^ 1;
        const int kn  = (step + 1) * TILE_K;
        const bool haveNext = (step + 1) < NSTEP;

        // (1) issue next tile's global loads (latency hidden under WMMA below)
        if (haveNext) {
            if (kn < H_DIM) {
                for (int q = 0; q < 4; ++q)
                    aReg[q] = *(const float4*)(hist_enc + (size_t)(r0 + aRow + 32*q) * H_DIM + kn + aK);
            }
            const float* bp = Wm + (size_t)(kn + bKh) * H_DIM + c0 + bCol;
            for (int i = 0; i < 16; ++i) bRegF[i] = bp[(size_t)i * H_DIM];
        }

        // (2) compute on current buffer: fragment loads + 8 WMMAs
        {
            v16h aF[2];
            for (int mt = 0; mt < 2; ++mt) {
                int lr = mGrp * 32 + mt * 16 + l15;
                const v8h lo = *(const v8h*)&ldsA[cur][lr * LDA + hi * 8];       // K = hi*8 + 0..7
                const v8h hp = *(const v8h*)&ldsA[cur][lr * LDA + 16 + hi * 8];  // K = 16 + hi*8 + 0..7
                v16h a;
                for (int i = 0; i < 8; ++i) { a[i] = lo[i]; a[8+i] = hp[i]; }
                aF[mt] = a;
            }
            for (int nt = 0; nt < 4; ++nt) {
                int lc = nGrp * 64 + nt * 16 + l15;
                const v8h lo = *(const v8h*)&ldsB[cur][lc * LDA + hi * 8];       // K = hi*8 + 0..7
                const v8h hp = *(const v8h*)&ldsB[cur][lc * LDA + 16 + hi * 8];  // K = 16 + hi*8 + 0..7
                v16h b;
                for (int i = 0; i < 8; ++i) { b[i] = lo[i]; b[8+i] = hp[i]; }
                acc[0][nt] = __builtin_amdgcn_wmma_f32_16x16x32_f16(
                    false, aF[0], false, b, (short)0, acc[0][nt], false, false);
                acc[1][nt] = __builtin_amdgcn_wmma_f32_16x16x32_f16(
                    false, aF[1], false, b, (short)0, acc[1][nt], false, false);
            }
        }

        // (3) convert + store next tile into the other buffer
        if (haveNext) {
            if (kn < H_DIM) {
                for (int q = 0; q < 4; ++q) {
                    v4h pk; pk[0] = (_Float16)aReg[q].x; pk[1] = (_Float16)aReg[q].y;
                            pk[2] = (_Float16)aReg[q].z; pk[3] = (_Float16)aReg[q].w;
                    *(v4h*)&ldsA[nxt][(aRow + 32*q) * LDA + aK] = pk;
                }
            } else {
                // emb part: relu(rel @ Ws + bs) computed on the fly (Ws/bs are L0/L2-hot)
                const int j = (kn - H_DIM) + aK;
                const float4 w0 = *(const float4*)(Ws + j);
                const float4 w1 = *(const float4*)(Ws + E_DIM + j);
                const float4 bv = *(const float4*)(bs + j);
                for (int q = 0; q < 4; ++q) {
                    int lr = aRow + 32*q;
                    float rx = relS[2*lr+0], ry = relS[2*lr+1];
                    v4h pk;
                    pk[0] = (_Float16)fmaxf(rx*w0.x + ry*w1.x + bv.x, 0.0f);
                    pk[1] = (_Float16)fmaxf(rx*w0.y + ry*w1.y + bv.y, 0.0f);
                    pk[2] = (_Float16)fmaxf(rx*w0.z + ry*w1.z + bv.z, 0.0f);
                    pk[3] = (_Float16)fmaxf(rx*w0.w + ry*w1.w + bv.w, 0.0f);
                    *(v4h*)&ldsA[nxt][lr * LDA + aK] = pk;
                }
            }
            for (int i2 = 0; i2 < 4; ++i2) {
                v4h pk;
                for (int j = 0; j < 4; ++j) pk[j] = (_Float16)bRegF[i2*4 + j];
                *(v4h*)&ldsB[nxt][bCol * LDA + bKh + i2*4] = pk;
            }
        }
        __syncthreads();
    }

    // ---- bias + ReLU, stage pooled tile in LDS ----
    for (int nt = 0; nt < 4; ++nt) {
        int lc = nGrp * 64 + nt * 16 + l15;
        float bias = bm[c0 + lc];
        for (int mt = 0; mt < 2; ++mt) {
            int rbase = mGrp * 32 + mt * 16 + hi * 8;  // C layout: vgpr v -> row v (+8 for hi lanes)
            for (int v = 0; v < 8; ++v) {
                float x = fmaxf(acc[mt][nt][v] + bias, 0.0f);
                pooled[(rbase + v) * LDP + lc] = x;
            }
        }
    }
    __syncthreads();

    // ---- segment max over rows; combine with uint atomicMax (values >= 0) ----
    {
        int col    = tid & 127;
        int rs     = (tid >> 7) * 64;
        int colAbs = c0 + col;
        float m  = pooled[rs * LDP + col];
        int curg = (r0 + rs) / GROUPS;
        for (int r = rs + 1; r < rs + 64; ++r) {
            int g   = (r0 + r) / GROUPS;   // uniform across threads -> no divergence
            float v = pooled[r * LDP + col];
            if (g != curg) {
                atomicMax((unsigned int*)&out[(size_t)curg * H_DIM + colAbs], __float_as_uint(m));
                curg = g;
                m = v;
            } else {
                m = fmaxf(m, v);
            }
        }
        atomicMax((unsigned int*)&out[(size_t)curg * H_DIM + colAbs], __float_as_uint(m));
    }
}

extern "C" void kernel_launch(void* const* d_in, const int* in_sizes, int n_in,
                              void* d_out, int out_size, void* d_ws, size_t ws_size,
                              hipStream_t stream) {
    const float* hist_enc = (const float*)d_in[0];
    const float* pos      = (const float*)d_in[1];
    const int*   sse      = (const int*)  d_in[2];
    const float* Ws       = (const float*)d_in[3];
    const float* bs       = (const float*)d_in[4];
    const float* Wm       = (const float*)d_in[5];
    const float* bm       = (const float*)d_in[6];
    float* out = (float*)d_out;

    init_out_kernel<<<(NGRP * H_DIM + 255) / 256, 256, 0, stream>>>(out);

    dim3 grid(H_DIM / TILE_N, N_TOT / TILE_M);   // (4, 800)
    pool_gemm_kernel<<<grid, 256, 0, stream>>>(hist_enc, pos, sse, Ws, bs, Wm, bm, out);
}
